// DecoderLayer_41016937677066
// MI455X (gfx1250) — matrix operand order repaired and checked by reading, not compiled, not run on hardware
//
#include <hip/hip_runtime.h>
#include <hip/hip_bf16.h>

// ---------------- problem constants ----------------
#define B_    2
#define S_    1024
#define D_    4096
#define NQ_   32
#define NKV_  8
#define HD_   128
#define FFN_  10928
#define MTOK  (B_ * S_)          // 2048 tokens
#define ATTN_MULT 0.08838834764831845f
#define MAX_ATTN  30.0f

typedef __attribute__((ext_vector_type(16))) __bf16 v16bf;
typedef __attribute__((ext_vector_type(8)))  __bf16 v8bf;
typedef __attribute__((ext_vector_type(4)))  __bf16 v4bf;
typedef __attribute__((ext_vector_type(8)))  float  v8f;
typedef __attribute__((ext_vector_type(4)))  float  v4f;

// native f32 -> bf16 (v_cvt)
__device__ __forceinline__ __bf16 f2bf(float f) { return (__bf16)f; }

// ---- WMMA fragment loaders (contiguous k order, same in A and B) ----
// A (16x32 bf16): lane&15 = row M; lanes 0-15 hold K 0-15,
// lanes 16-31 hold K 16-31.  Reads 2x16B contiguous -> ds_load_b128.
__device__ __forceinline__ v16bf load_a_frag(const __bf16* base, int stride, int lane) {
  const __bf16* p = base + (lane & 15) * stride + ((lane >> 4) << 4);
  v16bf a;
#pragma unroll
  for (int e = 0; e < 8; ++e) a[e] = p[e];
#pragma unroll
  for (int e = 0; e < 8; ++e) a[8 + e] = p[8 + e];
  return a;
}
// B (32x16 bf16) from a [N][K]-contiguous LDS tile: element e -> K=16*(lane>>4)+e,
// N=lane&15.  32B contiguous per lane -> 2x ds_load_b128.
__device__ __forceinline__ v16bf load_b_nk(const __bf16* base, int stride, int lane) {
  const __bf16* p = base + (lane & 15) * stride + ((lane >> 4) << 4);
  v16bf b;
#pragma unroll
  for (int e = 0; e < 16; ++e) b[e] = p[e];
  return b;
}

__device__ __forceinline__ v8f wmma_bf16(v16bf a, v16bf b, v8f c) {
  return __builtin_amdgcn_wmma_f32_16x16x32_bf16(false, a, false, b, (short)0, c,
                                                 false, false);
}

// =======================================================================
// Generic GEMM: C[M,N](f32) = A[M,K](f32) * W[K,N](f32), bf16 WMMA compute
// 128x128x64 block tile, 256 threads = 8 wave32, each wave 32x64.
// Weight tile is transposed into LDS ([n][k]) so B-frags are contiguous.
// =======================================================================
#define GTM 128
#define GTN 128
#define GTK 64
#define AP  (GTK + 8)    // 72 halfs = 144B pitch (16B aligned)

__global__ void __launch_bounds__(256)
gemm_bf16_kernel(const float* __restrict__ A, const float* __restrict__ W,
                 float* __restrict__ C, int M, int N, int K,
                 int lda, int ldb, int ldc) {
  __shared__ __align__(16) __bf16 As[2][GTM][AP];
  __shared__ __align__(16) __bf16 Bt[2][GTN][AP];   // transposed: [n][k]

  const int tid  = threadIdx.x;
  const int lane = tid & 31;
  const int wave = tid >> 5;          // wave32
  const int wm   = wave >> 1;         // 0..3 -> 32-row slab
  const int wn   = wave & 1;          // 0..1 -> 64-col slab
  const int bm   = blockIdx.y * GTM;
  const int bn   = blockIdx.x * GTN;

  v8f acc[2][4] = {};
  const int nk = (K + GTK - 1) / GTK;

  auto stage = [&](int buf, int k0) {
    // A tile 128x64 f32: 2048 float4 chunks, 8 per thread, coalesced.
#pragma unroll
    for (int i = 0; i < 8; ++i) {
      int idx = tid + i * 256;
      int r = idx >> 4, c4 = (idx & 15) << 2;
      int gr = bm + r, gc = k0 + c4;
      v4f v = {};
      if (gr < M && gc < K) v = *(const v4f*)&A[(long)gr * lda + gc];
      v4bf h;
#pragma unroll
      for (int j = 0; j < 4; ++j) h[j] = f2bf(v[j]);
      *(v4bf*)&As[buf][r][c4] = h;
    }
    // W tile 64x128 f32, transposed into Bt via 4x4 register blocks:
    // 512 blocks, 2 per thread; 4 coalesced global b128 loads + 4 ds_store_b64.
#pragma unroll
    for (int i = 0; i < 2; ++i) {
      int bi  = tid + i * 256;
      int kb4 = (bi >> 5) << 2;       // k offset 0..60
      int nb4 = (bi & 31) << 2;       // n offset 0..124
      v4f rowv[4];
#pragma unroll
      for (int j = 0; j < 4; ++j) {
        int gr = k0 + kb4 + j, gc = bn + nb4;
        rowv[j] = {};
        if (gr < K && gc < N) rowv[j] = *(const v4f*)&W[(long)gr * ldb + gc];
      }
#pragma unroll
      for (int j = 0; j < 4; ++j) {   // j = n offset within block
        v4bf h;
#pragma unroll
        for (int q = 0; q < 4; ++q) h[q] = f2bf(rowv[q][j]);
        *(v4bf*)&Bt[buf][nb4 + j][kb4] = h;
      }
    }
  };

  stage(0, 0);
  __syncthreads();
  for (int kt = 0; kt < nk; ++kt) {
    int buf = kt & 1;
    if (kt + 1 < nk) {
      if (kt + 2 < nk)  // hint next-next weight tile -> global_prefetch_b8
        __builtin_prefetch(&W[(long)(kt + 2) * GTK * ldb + bn + (tid & 127)], 0, 1);
      stage(buf ^ 1, (kt + 1) * GTK);
    }
#pragma unroll
    for (int kk = 0; kk < GTK / 32; ++kk) {
      v16bf af[2], bfv[4];
#pragma unroll
      for (int mt = 0; mt < 2; ++mt)
        af[mt] = load_a_frag(&As[buf][wm * 32 + mt * 16][kk * 32], AP, lane);
#pragma unroll
      for (int nt = 0; nt < 4; ++nt)
        bfv[nt] = load_b_nk(&Bt[buf][wn * 64 + nt * 16][kk * 32], AP, lane);
#pragma unroll
      for (int mt = 0; mt < 2; ++mt)
#pragma unroll
        for (int nt = 0; nt < 4; ++nt)
          acc[mt][nt] = wmma_bf16(af[mt], bfv[nt], acc[mt][nt]);
    }
    __syncthreads();
  }

  const int rhi = (lane >> 4) * 8;
  const int cn  = lane & 15;
#pragma unroll
  for (int mt = 0; mt < 2; ++mt)
#pragma unroll
    for (int nt = 0; nt < 4; ++nt)
#pragma unroll
      for (int vg = 0; vg < 8; ++vg) {
        int row = bm + wm * 32 + mt * 16 + vg + rhi;
        int col = bn + wn * 64 + nt * 16 + cn;
        if (row < M && col < N) C[(long)row * ldc + col] = acc[mt][nt][vg];
      }
}

// =======================================================================
// RMSNorm: out = x * rsqrt(mean(x^2)+eps) * scale      (one row / block)
// =======================================================================
__global__ void __launch_bounds__(256)
rmsnorm_kernel(const float* __restrict__ x, const float* __restrict__ scale,
               float* __restrict__ out, int D) {
  const int row = blockIdx.x;
  const int tid = threadIdx.x;
  const float* xr = x + (long)row * D;
  float ss = 0.f;
  for (int i = tid; i < D; i += 256) { float v = xr[i]; ss += v * v; }
#pragma unroll
  for (int off = 16; off > 0; off >>= 1) ss += __shfl_xor(ss, off, 32);
  __shared__ float red[8];
  if ((tid & 31) == 0) red[tid >> 5] = ss;
  __syncthreads();
  float tot = 0.f;
#pragma unroll
  for (int w = 0; w < 8; ++w) tot += red[w];
  float inv = rsqrtf(tot / (float)D + 1e-5f);
  for (int i = tid; i < D; i += 256) out[(long)row * D + i] = xr[i] * inv * scale[i];
}

// out = base + rmsnorm(t) * scale   (residual + post-norm)
__global__ void __launch_bounds__(256)
add_rmsnorm_kernel(const float* __restrict__ base, const float* __restrict__ t,
                   const float* __restrict__ scale, float* __restrict__ out, int D) {
  const int row = blockIdx.x;
  const int tid = threadIdx.x;
  const float* tr = t + (long)row * D;
  float ss = 0.f;
  for (int i = tid; i < D; i += 256) { float v = tr[i]; ss += v * v; }
#pragma unroll
  for (int off = 16; off > 0; off >>= 1) ss += __shfl_xor(ss, off, 32);
  __shared__ float red[8];
  if ((tid & 31) == 0) red[tid >> 5] = ss;
  __syncthreads();
  float tot = 0.f;
#pragma unroll
  for (int w = 0; w < 8; ++w) tot += red[w];
  float inv = rsqrtf(tot / (float)D + 1e-5f);
  for (int i = tid; i < D; i += 256)
    out[(long)row * D + i] = base[(long)row * D + i] + tr[i] * inv * scale[i];
}

// =======================================================================
// RoPE (offset 0) + cast to bf16.  in/out: [token][H][HD]
// =======================================================================
__global__ void __launch_bounds__(256)
rope_cast_kernel(const float* __restrict__ in, __bf16* __restrict__ out, int H) {
  const int token = blockIdx.x;
  const int s = token & (S_ - 1);
  const int n = H * HD_;
  const float* src = in + (long)token * n;
  __bf16* dst = out + (long)token * n;
  const float k = 0.14391156f;  // ln(10000)/64
  for (int idx = threadIdx.x; idx < n; idx += 256) {
    int d = idx & (HD_ - 1);
    int j = d & 63;
    float phase = (float)s * __expf(-k * (float)j);
    float sn, c;
    __sincosf(phase, &sn, &c);
    float xv = src[idx];
    float xr = (d < 64) ? -src[idx + 64] : src[idx - 64];
    dst[idx] = f2bf(xv * c + xr * sn);
  }
}

__global__ void __launch_bounds__(256)
cast_bf16_kernel(const float* __restrict__ in, __bf16* __restrict__ out, int n) {
  int i = blockIdx.x * 256 + threadIdx.x;
  if (i < n) out[i] = f2bf(in[i]);
}

__global__ void __launch_bounds__(256)
gelu_mul_kernel(const float* __restrict__ g, const float* __restrict__ v,
                float* __restrict__ out, int n) {
  int i = blockIdx.x * 256 + threadIdx.x;
  if (i < n) {
    float x = g[i];
    float t = tanhf(0.7978845608f * (x + 0.044715f * x * x * x));
    out[i] = 0.5f * x * (1.f + t) * v[i];
  }
}

// =======================================================================
// Flash attention with GQA, softcap, causal mask.  WMMA for QK^T and PV.
// Block: 128 q-rows of one (b, head); 8 waves, wave w owns q rows w*16..+15.
// =======================================================================
#define AQ 128
#define AK 64
#define QP (HD_ + 8)   // 136 halfs = 272B (16B aligned)
#define KP (HD_ + 8)
#define VP (AK + 8)    // 72 halfs = 144B (16B aligned)
#define PP (AK + 8)

__global__ void __launch_bounds__(256)
flash_attn_kernel(const __bf16* __restrict__ Q, const __bf16* __restrict__ K,
                  const __bf16* __restrict__ V, float* __restrict__ O) {
  __shared__ __align__(16) __bf16 Qs[AQ][QP];
  __shared__ __align__(16) __bf16 Ks[AK][KP];
  __shared__ __align__(16) __bf16 Vts[HD_][VP];  // V transposed: [hd][seq]
  __shared__ __align__(16) __bf16 Ps[AQ][PP];    // probs staging (per-wave rows)

  const int tid  = threadIdx.x;
  const int lane = tid & 31;
  const int wave = tid >> 5;
  const int qb   = blockIdx.x;      // 0..S/128-1
  const int head = blockIdx.y;      // 0..NQ-1
  const int b    = blockIdx.z;
  const int kvh  = head >> 2;       // G = 4
  const int q0   = qb * AQ;

  const long qbase  = (long)b * S_ * NQ_ * HD_ + (long)head * HD_;
  const long kvbase = (long)b * S_ * NKV_ * HD_ + (long)kvh * HD_;

  // stage Q block once: 2048 x 16B chunks, 8 per thread
#pragma unroll
  for (int i = 0; i < 8; ++i) {
    int idx = tid + i * 256;
    int r = idx >> 4, c8 = (idx & 15) << 3;
    *(v8bf*)&Qs[r][c8] =
        *(const v8bf*)&Q[qbase + (long)(q0 + r) * NQ_ * HD_ + c8];
  }

  v8f o_acc[8] = {};
  float mrow[8], lrow[8];
#pragma unroll
  for (int vg = 0; vg < 8; ++vg) { mrow[vg] = -1e30f; lrow[vg] = 0.f; }

  const int rhi = (lane >> 4) * 8;
  const int cn  = lane & 15;
  const int kb_end = 2 * (qb + 1);  // causal: only kv blocks <= q block

  for (int kb = 0; kb < kb_end; ++kb) {
    __syncthreads();
    // stage K (direct) and V (transposed): 1024 x 16B chunks, 4 per thread
#pragma unroll
    for (int i = 0; i < 4; ++i) {
      int idx = tid + i * 256;
      int r = idx >> 4, c8 = (idx & 15) << 3;
      long src = kvbase + (long)(kb * AK + r) * NKV_ * HD_ + c8;
      *(v8bf*)&Ks[r][c8] = *(const v8bf*)&K[src];
      v8bf vv = *(const v8bf*)&V[src];
#pragma unroll
      for (int j = 0; j < 8; ++j) Vts[c8 + j][r] = vv[j];
    }
    __syncthreads();

    // S = Q * K^T  (wave: 16 x 64)
    v8f s_acc[4] = {};
#pragma unroll
    for (int kk = 0; kk < HD_ / 32; ++kk) {
      v16bf a = load_a_frag(&Qs[wave * 16][kk * 32], QP, lane);
#pragma unroll
      for (int nt = 0; nt < 4; ++nt) {
        v16bf bb = load_b_nk(&Ks[nt * 16][kk * 32], KP, lane);
        s_acc[nt] = wmma_bf16(a, bb, s_acc[nt]);
      }
    }

    // softcap + causal mask + online softmax
#pragma unroll
    for (int vg = 0; vg < 8; ++vg) {
      int qi = q0 + wave * 16 + vg + rhi;
      float m_old = mrow[vg];
      float rmax = -1e30f;
      float xs[4];
#pragma unroll
      for (int nt = 0; nt < 4; ++nt) {
        float xv = s_acc[nt][vg] * ATTN_MULT;
        xv = MAX_ATTN * tanhf(xv * (1.f / MAX_ATTN));
        int ki = kb * AK + nt * 16 + cn;
        if (ki > qi) xv = -1e30f;
        xs[nt] = xv;
        rmax = fmaxf(rmax, xv);
      }
#pragma unroll
      for (int off = 8; off > 0; off >>= 1)
        rmax = fmaxf(rmax, __shfl_xor(rmax, off, 16));
      float m_new = fmaxf(m_old, rmax);
      float alpha = __expf(m_old - m_new);
      float rs = 0.f;
#pragma unroll
      for (int nt = 0; nt < 4; ++nt) {
        float p = __expf(xs[nt] - m_new);
        rs += p;
        Ps[wave * 16 + vg + rhi][nt * 16 + cn] = f2bf(p);
      }
#pragma unroll
      for (int off = 8; off > 0; off >>= 1) rs += __shfl_xor(rs, off, 16);
      lrow[vg] = lrow[vg] * alpha + rs;
      mrow[vg] = m_new;
#pragma unroll
      for (int nt = 0; nt < 8; ++nt) o_acc[nt][vg] *= alpha;
    }

    // O += P * V   (K-dim = 64 seq positions)
#pragma unroll
    for (int kk = 0; kk < AK / 32; ++kk) {
      v16bf a = load_a_frag(&Ps[wave * 16][kk * 32], PP, lane);
#pragma unroll
      for (int nt = 0; nt < 8; ++nt) {
        v16bf bb = load_b_nk(&Vts[nt * 16][kk * 32], VP, lane);
        o_acc[nt] = wmma_bf16(a, bb, o_acc[nt]);
      }
    }
  }

  // normalize and write O (f32, [b][s][head][hd])
#pragma unroll
  for (int vg = 0; vg < 8; ++vg) {
    float invl = 1.f / lrow[vg];
    int q = q0 + wave * 16 + vg + rhi;
#pragma unroll
    for (int nt = 0; nt < 8; ++nt)
      O[qbase + (long)q * NQ_ * HD_ + nt * 16 + cn] = o_acc[nt][vg] * invl;
  }
}

// =======================================================================
// Orchestration
// =======================================================================
extern "C" void kernel_launch(void* const* d_in, const int* in_sizes, int n_in,
                              void* d_out, int out_size, void* d_ws, size_t ws_size,
                              hipStream_t stream) {
  (void)in_sizes; (void)n_in; (void)out_size; (void)ws_size;
  const float* x           = (const float*)d_in[0];
  // d_in[1] = causal mask -> handled analytically
  const float* s_pre_attn  = (const float*)d_in[2];
  const float* s_post_attn = (const float*)d_in[3];
  const float* s_pre_mlp   = (const float*)d_in[4];
  const float* s_post_mlp  = (const float*)d_in[5];
  const float* wq   = (const float*)d_in[6];
  const float* wk   = (const float*)d_in[7];
  const float* wv   = (const float*)d_in[8];
  const float* wo   = (const float*)d_in[9];
  const float* wg   = (const float*)d_in[10];
  const float* wval = (const float*)d_in[11];
  const float* wout = (const float*)d_in[12];
  float* out = (float*)d_out;

  char* ws = (char*)d_ws;
  size_t off = 0;
  auto alloc = [&](size_t bytes) -> void* {
    void* p = ws + off;
    off += (bytes + 255) & ~(size_t)255;
    return p;
  };
  float*  hn   = (float*) alloc((size_t)MTOK * D_ * 4);
  float*  qf   = (float*) alloc((size_t)MTOK * NQ_ * HD_ * 4);
  float*  kf   = (float*) alloc((size_t)MTOK * NKV_ * HD_ * 4);
  float*  vf   = (float*) alloc((size_t)MTOK * NKV_ * HD_ * 4);
  __bf16* qb16 = (__bf16*)alloc((size_t)MTOK * NQ_ * HD_ * 2);
  __bf16* kb16 = (__bf16*)alloc((size_t)MTOK * NKV_ * HD_ * 2);
  __bf16* vb16 = (__bf16*)alloc((size_t)MTOK * NKV_ * HD_ * 2);
  float*  attn = (float*) alloc((size_t)MTOK * D_ * 4);
  float*  gate = (float*) alloc((size_t)MTOK * FFN_ * 4);
  float*  val  = (float*) alloc((size_t)MTOK * FFN_ * 4);
  float*  hattn = qf;    // reuse: qf dead after rope_cast
  float*  hmlp  = attn;  // reuse: attn dead after o-proj

  dim3 blk(256);

  // ---- attention sublayer ----
  rmsnorm_kernel<<<MTOK, blk, 0, stream>>>(x, s_pre_attn, hn, D_);
  gemm_bf16_kernel<<<dim3((NQ_*HD_)/GTN, MTOK/GTM), blk, 0, stream>>>(
      hn, wq, qf, MTOK, NQ_*HD_, D_, D_, NQ_*HD_, NQ_*HD_);
  gemm_bf16_kernel<<<dim3((NKV_*HD_)/GTN, MTOK/GTM), blk, 0, stream>>>(
      hn, wk, kf, MTOK, NKV_*HD_, D_, D_, NKV_*HD_, NKV_*HD_);
  gemm_bf16_kernel<<<dim3((NKV_*HD_)/GTN, MTOK/GTM), blk, 0, stream>>>(
      hn, wv, vf, MTOK, NKV_*HD_, D_, D_, NKV_*HD_, NKV_*HD_);
  rope_cast_kernel<<<MTOK, blk, 0, stream>>>(qf, qb16, NQ_);
  rope_cast_kernel<<<MTOK, blk, 0, stream>>>(kf, kb16, NKV_);
  cast_bf16_kernel<<<(MTOK * NKV_ * HD_) / 256, blk, 0, stream>>>(
      vf, vb16, MTOK * NKV_ * HD_);
  flash_attn_kernel<<<dim3(S_/AQ, NQ_, B_), blk, 0, stream>>>(qb16, kb16, vb16, attn);
  gemm_bf16_kernel<<<dim3(D_/GTN, MTOK/GTM), blk, 0, stream>>>(
      attn, wo, hattn, MTOK, D_, NQ_*HD_, NQ_*HD_, D_, D_);
  add_rmsnorm_kernel<<<MTOK, blk, 0, stream>>>(x, hattn, s_post_attn, out, D_);

  // ---- gated MLP sublayer ----
  rmsnorm_kernel<<<MTOK, blk, 0, stream>>>(out, s_pre_mlp, hn, D_);
  const int gN = (FFN_ + GTN - 1) / GTN;  // 86
  gemm_bf16_kernel<<<dim3(gN, MTOK/GTM), blk, 0, stream>>>(
      hn, wg, gate, MTOK, FFN_, D_, D_, FFN_, FFN_);
  gemm_bf16_kernel<<<dim3(gN, MTOK/GTM), blk, 0, stream>>>(
      hn, wval, val, MTOK, FFN_, D_, D_, FFN_, FFN_);
  const int nmlp = MTOK * FFN_;
  gelu_mul_kernel<<<(nmlp + 255) / 256, blk, 0, stream>>>(gate, val, gate, nmlp);
  gemm_bf16_kernel<<<dim3(D_/GTN, MTOK/GTM), blk, 0, stream>>>(
      gate, wout, hmlp, MTOK, D_, FFN_, FFN_, D_, D_);
  add_rmsnorm_kernel<<<MTOK, blk, 0, stream>>>(out, hmlp, s_post_mlp, out, D_);
}
